// VQVAE_5652176961760
// MI455X (gfx1250) — matrix operand order repaired and checked by reading, not compiled, not run on hardware
//
#include <hip/hip_runtime.h>

// ---------------------------------------------------------------------------
// CDNA5 (gfx1250) VQ-VAE forward: all GEMM-shaped work on v_wmma_f32_16x16x32_bf16.
// Activations: bf16 NHWC (channel-contiguous) so im2col B-fragments are plain
// pairs of contiguous 16B loads. fp32 accumulation via WMMA C/D.
// All layer shapes are compile-time template parameters so tile decode is
// shift/mask only (no SALU division) and the K-loops fully unroll into dense
// global_load_b128 + v_wmma streams.
// ---------------------------------------------------------------------------

typedef __bf16 bf16;
typedef __attribute__((ext_vector_type(16))) __bf16 v16bf;
typedef __attribute__((ext_vector_type(8)))  __bf16 v8bf;
typedef __attribute__((ext_vector_type(8)))  float  v8f;

#define DEVINL static __device__ __forceinline__

DEVINL v16bf cat8(v8bf lo, v8bf hi) {
  v16bf r;
#pragma unroll
  for (int i = 0; i < 8; ++i) { r[i] = lo[i]; r[i + 8] = hi[i]; }
  return r;
}

DEVINL v16bf zero16() {
  v16bf r;
#pragma unroll
  for (int i = 0; i < 16; ++i) r[i] = (bf16)0.0f;
  return r;
}

// bf16 relu via sign-bit test (negatives -> +0)
DEVINL v16bf relu16(v16bf x) {
  v16bf r;
#pragma unroll
  for (int i = 0; i < 16; ++i) {
    unsigned short u = __builtin_bit_cast(unsigned short, (bf16)x[i]);
    u = (unsigned short)((u & 0x8000u) ? 0u : u);
    r[i] = __builtin_bit_cast(bf16, u);
  }
  return r;
}

DEVINL v8f wmma_bf16(v16bf a, v16bf b, v8f c) {
  // 8 args: (neg_a, A, neg_b, B, c_mod, C, reuse_a, reuse_b)
  return __builtin_amdgcn_wmma_f32_16x16x32_bf16(false, a, false, b, (short)0, c,
                                                 false, false);
}

// ---------------------------------------------------------------------------
// Weight packing: fp32 (arbitrary strides: OIHW or transposed IOHW) ->
// bf16 tiles [g = ky*KW+kx][OC][IC], IC contiguous (matches A-fragment loads).
// ---------------------------------------------------------------------------
__global__ void pack_w(const float* __restrict__ w, bf16* __restrict__ out,
                       int OC, int IC, int KW, int soc, int sic, int sky, int skx,
                       int total) {
  int i = blockIdx.x * 256 + threadIdx.x;
  if (i >= total) return;
  int ic = i % IC;
  int t  = i / IC;
  int oc = t % OC;
  int g  = t / OC;
  int ky = g / KW, kx = g % KW;
  out[i] = (bf16)w[(size_t)oc * soc + (size_t)ic * sic + ky * sky + kx * skx];
}

__global__ void pack_cb(const float* __restrict__ cb, bf16* __restrict__ cbbf,
                        float* __restrict__ cn2) {
  int e = blockIdx.x * blockDim.x + threadIdx.x;
  if (e >= 512) return;
  float s = 0.f;
  for (int j = 0; j < 64; ++j) {
    bf16 q = (bf16)cb[e * 64 + j];
    cbbf[e * 64 + j] = q;
    float v = (float)q;
    s += v * v;
  }
  cn2[e] = s;
}

__global__ void zero_f(float* __restrict__ p, int n) {
  int i = blockIdx.x * blockDim.x + threadIdx.x;
  if (i < n) p[i] = 0.f;
}

// ---------------------------------------------------------------------------
// Encoder conv1: 3->64, 4x4, stride2, pad1  (IC=3: direct VALU, writes bf16 NHWC)
// out [64][64][64][64ch], pre-activation (ReLU fused into next layer's load).
// ---------------------------------------------------------------------------
__global__ void __launch_bounds__(256)
enc_conv1(const float* __restrict__ x, const float* __restrict__ w,
          const float* __restrict__ bias, bf16* __restrict__ out) {
  int tid = blockIdx.x * 256 + threadIdx.x;            // 64*64*64*64
  int oc = tid & 63;
  int ox = (tid >> 6) & 63;
  int oy = (tid >> 12) & 63;
  int b  = tid >> 18;
  float acc = bias[oc];
#pragma unroll
  for (int ky = 0; ky < 4; ++ky) {
    int iy = oy * 2 - 1 + ky;
    if (iy < 0 || iy >= 128) continue;
#pragma unroll
    for (int kx = 0; kx < 4; ++kx) {
      int ix = ox * 2 - 1 + kx;
      if (ix < 0 || ix >= 128) continue;
#pragma unroll
      for (int ic = 0; ic < 3; ++ic) {
        acc += x[((size_t)(b * 3 + ic) * 128 + iy) * 128 + ix] *
               w[((oc * 3 + ic) * 4 + ky) * 4 + kx];
      }
    }
  }
  out[tid] = (bf16)acc;
}

// ---------------------------------------------------------------------------
// Generic WMMA conv (implicit GEMM), fully compile-time shaped.
// One wave = 16 OCs x 16 pixels (same oy). blockDim = OC*2 (OC/16 waves),
// grid = B*OUTH*(OUTW/16). Fused: input ReLU, residual add, bias.
// OUTW/16 and OUTH are powers of two -> decode is shift/mask only.
// ---------------------------------------------------------------------------
template <int IC, int OC, int KH, int KW, int STRIDE, int PAD,
          int INH, int INW, int OUTH, int OUTW, bool RELU, bool HAS_SKIP>
__global__ void __launch_bounds__(OC * 2)
conv_wmma(const bf16* __restrict__ in, const bf16* __restrict__ wp,
          const float* __restrict__ bias, const bf16* __restrict__ skip,
          bf16* __restrict__ out) {
  constexpr int TPR = OUTW / 16;   // tiles per row (2 or 4): power of two
  const int lane = threadIdx.x & 31;
  const int half = lane >> 4;      // K-half select (A: +8, B: +8)
  const int n    = lane & 15;      // pixel within tile / A-matrix row
  const int m0   = (threadIdx.x >> 5) << 4;

  int t = blockIdx.x;
  const int ox = (t & (TPR - 1)) * 16 + n;
  t >>= (TPR == 4 ? 2 : 1);
  const int oy = t & (OUTH - 1);
  const int b  = t / OUTH;         // OUTH power of two -> shift

  v8f acc;
#pragma unroll
  for (int i = 0; i < 8; ++i) acc[i] = 0.0f;

  constexpr int ICSTEPS = IC / 32;
#pragma unroll
  for (int ky = 0; ky < KH; ++ky) {
    const int iy = oy * STRIDE - PAD + ky;
    if (iy < 0 || iy >= INH) continue;             // uniform per block
#pragma unroll
    for (int kx = 0; kx < KW; ++kx) {
      const int ix  = ox * STRIDE - PAD + kx;
      const bool ib = (ix >= 0) && (ix < INW);     // per-lane zero pad
      const bf16* pix = in + (size_t)((b * INH + iy) * INW + ix) * IC + half * 8;
      const bf16* ap  = wp + (size_t)((ky * KW + kx) * OC + m0 + n) * IC + half * 8;
#pragma unroll
      for (int s = 0; s < ICSTEPS; ++s) {
        v16bf afrag = cat8(*(const v8bf*)(ap + s * 32),
                           *(const v8bf*)(ap + s * 32 + 16));
        v16bf bfrag = zero16();
        if (ib) bfrag = cat8(*(const v8bf*)(pix + s * 32), *(const v8bf*)(pix + s * 32 + 16));
        if (RELU) bfrag = relu16(bfrag);
        acc = wmma_bf16(afrag, bfrag, acc);
      }
    }
  }

  const size_t pix_out = (size_t)((b * OUTH + oy) * OUTW + ox);
  const int oc0 = m0 + half * 8;                   // C/D: row = v + 8*half
  const float* bp = bias + oc0;
  bf16* op = out + pix_out * OC + oc0;
  v8bf o;
  if (HAS_SKIP) {
    v8bf sk = *(const v8bf*)(skip + pix_out * OC + oc0);
#pragma unroll
    for (int v = 0; v < 8; ++v) o[v] = (bf16)(acc[v] + bp[v] + (float)sk[v]);
  } else {
#pragma unroll
    for (int v = 0; v < 8; ++v) o[v] = (bf16)(acc[v] + bp[v]);
  }
  *(v8bf*)op = o;
}

// ---------------------------------------------------------------------------
// Transposed conv 4x4 s2 p1 via WMMA: out(oy,ox) gets exactly 2x2 taps.
// N-tile = 16 same-parity output pixels (ox = 2*(qx0+n)+px). K = 4*IC.
// Weight groups g = ky*4+kx packed from [IC][OC][4][4].
// Fixed shape: 128ch 32x32 -> 64ch 64x64 with fused input ReLU.
// ---------------------------------------------------------------------------
template <int IC, int OC, int INH, int INW, int OUTH, int OUTW>
__global__ void __launch_bounds__(OC * 2)
convt_wmma(const bf16* __restrict__ in, const bf16* __restrict__ wp,
           const float* __restrict__ bias, bf16* __restrict__ out) {
  constexpr int TPR = OUTW / 16;   // 4
  const int lane = threadIdx.x & 31;
  const int half = lane >> 4;
  const int n    = lane & 15;
  const int m0   = (threadIdx.x >> 5) << 4;

  int t = blockIdx.x;
  const int sub = t & (TPR - 1);
  t >>= 2;
  const int oy = t & (OUTH - 1);
  const int b  = t / OUTH;

  const int px = sub & 1;
  const int qx = ((sub >> 1) << 4) + n;
  const int ox = (qx << 1) + px;
  const int py = oy & 1;
  const int qy = oy >> 1;

  v8f acc;
#pragma unroll
  for (int i = 0; i < 8; ++i) acc[i] = 0.0f;

  constexpr int ICSTEPS = IC / 32;
#pragma unroll
  for (int ty = 0; ty < 2; ++ty) {
    const int iy = qy + py - ty;
    if (iy < 0 || iy >= INH) continue;             // uniform per block
    const int ky = 1 - py + 2 * ty;
#pragma unroll
    for (int tx = 0; tx < 2; ++tx) {
      const int ix  = qx + px - tx;
      const bool ib = (ix >= 0) && (ix < INW);
      const int kx = 1 - px + 2 * tx;
      const int g  = ky * 4 + kx;
      const bf16* pix = in + (size_t)((b * INH + iy) * INW + ix) * IC + half * 8;
      const bf16* ap  = wp + (size_t)(g * OC + m0 + n) * IC + half * 8;
#pragma unroll
      for (int s = 0; s < ICSTEPS; ++s) {
        v16bf afrag = cat8(*(const v8bf*)(ap + s * 32),
                           *(const v8bf*)(ap + s * 32 + 16));
        v16bf bfrag = zero16();
        if (ib) bfrag = cat8(*(const v8bf*)(pix + s * 32), *(const v8bf*)(pix + s * 32 + 16));
        bfrag = relu16(bfrag);                     // res-stack output ReLU
        acc = wmma_bf16(afrag, bfrag, acc);
      }
    }
  }

  const size_t pix_out = (size_t)((b * OUTH + oy) * OUTW + ox);
  const int oc0 = m0 + half * 8;
  const float* bp = bias + oc0;
  bf16* op = out + pix_out * OC + oc0;
  v8bf o;
#pragma unroll
  for (int v = 0; v < 8; ++v) o[v] = (bf16)(acc[v] + bp[v]);
  *(v8bf*)op = o;
}

// ---------------------------------------------------------------------------
// VQ argmin via WMMA: each wave owns 16 ze rows (A fragment, K=64 -> 2 steps),
// sweeps 32 codebook tiles of 16 codes. score = ||c||^2 - 2*ze.c (||ze||^2
// is row-constant, irrelevant for argmin). Half-wave shuffle-reduce argmin.
// ---------------------------------------------------------------------------
__global__ void __launch_bounds__(256)
vq_argmin(const bf16* __restrict__ ze, const bf16* __restrict__ cb,
          const float* __restrict__ cn2, int* __restrict__ idx) {
  const int lane = threadIdx.x & 31;
  const int half = lane >> 4;
  const int an   = lane & 15;
  const int wv   = threadIdx.x >> 5;
  const int r0   = blockIdx.x * 128 + wv * 16;

  const bf16* zp = ze + (size_t)(r0 + an) * 64;
  v16bf a0 = cat8(*(const v8bf*)(zp + half * 8),      *(const v8bf*)(zp + half * 8 + 16));
  v16bf a1 = cat8(*(const v8bf*)(zp + 32 + half * 8), *(const v8bf*)(zp + 32 + half * 8 + 16));

  float best[8];
  int   bidx[8];
#pragma unroll
  for (int v = 0; v < 8; ++v) { best[v] = 3.4e38f; bidx[v] = 0; }

  for (int ct = 0; ct < 32; ++ct) {
    const int code = ct * 16 + an;                 // B-fragment column N = an
    const bf16* cp = cb + (size_t)code * 64 + half * 8;
    v16bf b0 = cat8(*(const v8bf*)cp, *(const v8bf*)(cp + 16));
    v16bf b1 = cat8(*(const v8bf*)(cp + 32), *(const v8bf*)(cp + 48));
    v8f acc;
#pragma unroll
    for (int i = 0; i < 8; ++i) acc[i] = 0.0f;
    acc = wmma_bf16(a0, b0, acc);
    acc = wmma_bf16(a1, b1, acc);
    const float cn = cn2[code];
#pragma unroll
    for (int v = 0; v < 8; ++v) {
      float s = cn - 2.0f * acc[v];                // row r0 + v + 8*half
      if (s < best[v]) { best[v] = s; bidx[v] = code; }
    }
  }

  // Min-reduce over the 16 lanes of each half (they share rows, differ in code).
  for (int m = 1; m < 16; m <<= 1) {
#pragma unroll
    for (int v = 0; v < 8; ++v) {
      float ov = __shfl_xor(best[v], m, 32);
      int   oi = __shfl_xor(bidx[v], m, 32);
      if (ov < best[v] || (ov == best[v] && oi < bidx[v])) {
        best[v] = ov;
        bidx[v] = oi;
      }
    }
  }
  if (an == 0) {
#pragma unroll
    for (int v = 0; v < 8; ++v) idx[r0 + half * 8 + v] = bidx[v];
  }
}

// Gather zq rows, accumulate sum((zq-ze)^2) for both losses.
__global__ void __launch_bounds__(256)
vq_gather_loss(const bf16* __restrict__ ze, const bf16* __restrict__ cb,
               const int* __restrict__ idx, bf16* __restrict__ zq,
               float* __restrict__ sum, int N) {
  int row = blockIdx.x * 256 + threadIdx.x;
  float s = 0.f;
  if (row < N) {
    int e = idx[row];
    const bf16* zr = ze + (size_t)row * 64;
    const bf16* cr = cb + (size_t)e * 64;
    bf16* qr = zq + (size_t)row * 64;
    for (int j = 0; j < 64; ++j) {
      bf16 q = cr[j];
      qr[j] = q;
      float d = (float)q - (float)zr[j];
      s += d * d;
    }
  }
#pragma unroll
  for (int m = 1; m < 32; m <<= 1) s += __shfl_xor(s, m, 32);
  if ((threadIdx.x & 31) == 0) atomicAdd(sum, s);
}

__global__ void finalize_losses(const float* __restrict__ sum, float* __restrict__ out) {
  float S = sum[0] * (1.0f / 4194304.0f);          // mean over 65536*64
  out[0] = S;                                      // L_emb
  out[1] = 0.25f * S;                              // l_commit = BETA * L_emb
}

// ---------------------------------------------------------------------------
// Decoder convT2: 64->3, 4x4 s2 p1, 64x64 -> 128x128. OC=3 (not WMMA shaped):
// direct kernel, fuses the preceding ReLU, writes fp32 NCHW to d_out.
// ---------------------------------------------------------------------------
__global__ void __launch_bounds__(256)
dec_convt2(const bf16* __restrict__ in, const float* __restrict__ w,
           const float* __restrict__ bias, float* __restrict__ out) {
  int tid = blockIdx.x * 256 + threadIdx.x;        // 64*3*128*128
  int ox = tid & 127;
  int oy = (tid >> 7) & 127;
  int bc = tid >> 14;
  int c = bc % 3;
  int b = bc / 3;
  int px = ox & 1, qx = ox >> 1;
  int py = oy & 1, qy = oy >> 1;
  float acc = bias[c];
#pragma unroll
  for (int ty = 0; ty < 2; ++ty) {
    int iy = qy + py - ty;
    if (iy < 0 || iy >= 64) continue;
    int ky = 1 - py + 2 * ty;
#pragma unroll
    for (int tx = 0; tx < 2; ++tx) {
      int ix = qx + px - tx;
      if (ix < 0 || ix >= 64) continue;
      int kx = 1 - px + 2 * tx;
      const bf16* pr = in + (size_t)((b * 64 + iy) * 64 + ix) * 64;
      for (int ic = 0; ic < 64; ++ic) {
        float v = (float)pr[ic];
        v = v > 0.f ? v : 0.f;                     // fused ReLU of convT1 output
        acc += v * w[((ic * 3 + c) * 4 + ky) * 4 + kx];
      }
    }
  }
  out[tid] = acc;
}

// ---------------------------------------------------------------------------
// Host: orchestrate the whole forward pass on `stream`.
// ---------------------------------------------------------------------------
extern "C" void kernel_launch(void* const* d_in, const int* in_sizes, int n_in,
                              void* d_out, int out_size, void* d_ws, size_t ws_size,
                              hipStream_t stream) {
  (void)in_sizes; (void)n_in; (void)out_size; (void)ws_size;

  const float* x        = (const float*)d_in[0];
  const float* enc_w1   = (const float*)d_in[1];
  const float* enc_b1   = (const float*)d_in[2];
  const float* enc_w2   = (const float*)d_in[3];
  const float* enc_b2   = (const float*)d_in[4];
  const float* enc_w3   = (const float*)d_in[5];
  const float* enc_b3   = (const float*)d_in[6];
  const float* enc_rw3  = (const float*)d_in[7];
  const float* enc_rb3  = (const float*)d_in[8];
  const float* enc_rw1  = (const float*)d_in[9];
  const float* enc_rb1  = (const float*)d_in[10];
  const float* last_w   = (const float*)d_in[11];
  const float* last_b   = (const float*)d_in[12];
  const float* dec_w1   = (const float*)d_in[13];
  const float* dec_b1   = (const float*)d_in[14];
  const float* dec_rw3  = (const float*)d_in[15];
  const float* dec_rb3  = (const float*)d_in[16];
  const float* dec_rw1  = (const float*)d_in[17];
  const float* dec_rb1  = (const float*)d_in[18];
  const float* dec_tw1  = (const float*)d_in[19];
  const float* dec_tb1  = (const float*)d_in[20];
  const float* dec_tw2  = (const float*)d_in[21];
  const float* dec_tb2  = (const float*)d_in[22];
  const float* codebook = (const float*)d_in[23];
  float* out = (float*)d_out;

  // ---- workspace carve-out (256B aligned) ----
  char* ws = (char*)d_ws;
  size_t off = 0;
  auto take = [&](size_t bytes) -> char* {
    char* p = ws + off;
    off += (bytes + 255) & ~(size_t)255;
    return p;
  };
  bf16* WP2    = (bf16*)take((size_t)16 * 128 * 64 * 2);
  bf16* WP3    = (bf16*)take((size_t)9 * 128 * 128 * 2);
  bf16* WPER3  = (bf16*)take((size_t)2 * 9 * 32 * 128 * 2);
  bf16* WPER1  = (bf16*)take((size_t)2 * 128 * 32 * 2);
  bf16* WPLAST = (bf16*)take((size_t)9 * 64 * 128 * 2);
  bf16* WPD1   = (bf16*)take((size_t)9 * 128 * 64 * 2);
  bf16* WPDR3  = (bf16*)take((size_t)2 * 9 * 32 * 128 * 2);
  bf16* WPDR1  = (bf16*)take((size_t)2 * 128 * 32 * 2);
  bf16* WPT1   = (bf16*)take((size_t)16 * 64 * 128 * 2);
  bf16* CBBF   = (bf16*)take((size_t)512 * 64 * 2);
  float* CN2   = (float*)take((size_t)512 * 4);
  int*  IDX    = (int*)take((size_t)65536 * 4);
  float* SUM   = (float*)take(256);
  bf16* BUFA   = (bf16*)take((size_t)64 * 64 * 64 * 64 * 2);   // conv1 out; later convT1 out
  bf16* BUFB   = (bf16*)take((size_t)64 * 32 * 32 * 128 * 2);  // conv2 out; later dec x
  bf16* BUFC   = (bf16*)take((size_t)64 * 32 * 32 * 128 * 2);  // conv3 out / enc res x
  bf16* BUFD   = (bf16*)take((size_t)64 * 32 * 32 * 128 * 2);  // res ping-pong
  bf16* HBUF   = (bf16*)take((size_t)64 * 32 * 32 * 32 * 2);   // res hidden (32ch)
  bf16* ZE     = (bf16*)take((size_t)64 * 32 * 32 * 64 * 2);
  bf16* ZQ     = (bf16*)take((size_t)64 * 32 * 32 * 64 * 2);

  // ---- weight packing ----
  auto packN = [&](const float* w, bf16* dst, int OC, int IC, int KH, int KW,
                   int soc, int sic, int sky, int skx) {
    int total = KH * KW * OC * IC;
    pack_w<<<dim3((total + 255) / 256), dim3(256), 0, stream>>>(
        w, dst, OC, IC, KW, soc, sic, sky, skx, total);
  };
  packN(enc_w2, WP2, 128, 64, 4, 4, 64 * 16, 16, 4, 1);       // OIHW
  packN(enc_w3, WP3, 128, 128, 3, 3, 128 * 9, 9, 3, 1);
  for (int l = 0; l < 2; ++l) {
    packN(enc_rw3 + (size_t)l * 32 * 128 * 9, WPER3 + (size_t)l * 9 * 32 * 128,
          32, 128, 3, 3, 128 * 9, 9, 3, 1);
    packN(enc_rw1 + (size_t)l * 128 * 32, WPER1 + (size_t)l * 128 * 32,
          128, 32, 1, 1, 32, 1, 0, 0);
    packN(dec_rw3 + (size_t)l * 32 * 128 * 9, WPDR3 + (size_t)l * 9 * 32 * 128,
          32, 128, 3, 3, 128 * 9, 9, 3, 1);
    packN(dec_rw1 + (size_t)l * 128 * 32, WPDR1 + (size_t)l * 128 * 32,
          128, 32, 1, 1, 32, 1, 0, 0);
  }
  packN(last_w, WPLAST, 64, 128, 3, 3, 128 * 9, 9, 3, 1);
  packN(dec_w1, WPD1, 128, 64, 3, 3, 64 * 9, 9, 3, 1);
  packN(dec_tw1, WPT1, 64, 128, 4, 4, /*soc=*/16, /*sic=*/64 * 16, 4, 1);  // [IC][OC][4][4]
  pack_cb<<<dim3(2), dim3(256), 0, stream>>>(codebook, CBBF, CN2);
  zero_f<<<dim3(1), dim3(64), 0, stream>>>(SUM, 64);

  // ---- encoder ----
  enc_conv1<<<dim3(65536), dim3(256), 0, stream>>>(x, enc_w1, enc_b1, BUFA);
  // conv2: 64->128, 4x4 s2 p1, 64x64 -> 32x32, ReLU(in)
  conv_wmma<64, 128, 4, 4, 2, 1, 64, 64, 32, 32, true, false>
      <<<dim3(64 * 32 * 2), dim3(256), 0, stream>>>(BUFA, WP2, enc_b2, nullptr, BUFB);
  // conv3: 128->128, 3x3 s1 p1, ReLU(in)
  conv_wmma<128, 128, 3, 3, 1, 1, 32, 32, 32, 32, true, false>
      <<<dim3(64 * 32 * 2), dim3(256), 0, stream>>>(BUFB, WP3, enc_b3, nullptr, BUFC);
  // residual stack (pre-activation buffers; ReLU fused into conv inputs)
  conv_wmma<128, 32, 3, 3, 1, 1, 32, 32, 32, 32, true, false>
      <<<dim3(64 * 32 * 2), dim3(64), 0, stream>>>(BUFC, WPER3, enc_rb3, nullptr, HBUF);
  conv_wmma<32, 128, 1, 1, 1, 0, 32, 32, 32, 32, true, true>
      <<<dim3(64 * 32 * 2), dim3(256), 0, stream>>>(HBUF, WPER1, enc_rb1, BUFC, BUFD);
  conv_wmma<128, 32, 3, 3, 1, 1, 32, 32, 32, 32, true, false>
      <<<dim3(64 * 32 * 2), dim3(64), 0, stream>>>(
          BUFD, WPER3 + (size_t)9 * 32 * 128, enc_rb3 + 32, nullptr, HBUF);
  conv_wmma<32, 128, 1, 1, 1, 0, 32, 32, 32, 32, true, true>
      <<<dim3(64 * 32 * 2), dim3(256), 0, stream>>>(
          HBUF, WPER1 + (size_t)128 * 32, enc_rb1 + 128, BUFD, BUFC);
  // last: 128->64, 3x3 s1 p1, ReLU(in) (res-stack output relu)
  conv_wmma<128, 64, 3, 3, 1, 1, 32, 32, 32, 32, true, false>
      <<<dim3(64 * 32 * 2), dim3(128), 0, stream>>>(BUFC, WPLAST, last_b, nullptr, ZE);

  // ---- vector quantization ----
  vq_argmin<<<dim3(512), dim3(256), 0, stream>>>(ZE, CBBF, CN2, IDX);
  vq_gather_loss<<<dim3(256), dim3(256), 0, stream>>>(ZE, CBBF, IDX, ZQ, SUM, 65536);

  // ---- decoder ----
  conv_wmma<64, 128, 3, 3, 1, 1, 32, 32, 32, 32, false, false>
      <<<dim3(64 * 32 * 2), dim3(256), 0, stream>>>(ZQ, WPD1, dec_b1, nullptr, BUFB);
  conv_wmma<128, 32, 3, 3, 1, 1, 32, 32, 32, 32, true, false>
      <<<dim3(64 * 32 * 2), dim3(64), 0, stream>>>(BUFB, WPDR3, dec_rb3, nullptr, HBUF);
  conv_wmma<32, 128, 1, 1, 1, 0, 32, 32, 32, 32, true, true>
      <<<dim3(64 * 32 * 2), dim3(256), 0, stream>>>(HBUF, WPDR1, dec_rb1, BUFB, BUFD);
  conv_wmma<128, 32, 3, 3, 1, 1, 32, 32, 32, 32, true, false>
      <<<dim3(64 * 32 * 2), dim3(64), 0, stream>>>(
          BUFD, WPDR3 + (size_t)9 * 32 * 128, dec_rb3 + 32, nullptr, HBUF);
  conv_wmma<32, 128, 1, 1, 1, 0, 32, 32, 32, 32, true, true>
      <<<dim3(64 * 32 * 2), dim3(256), 0, stream>>>(
          HBUF, WPDR1 + (size_t)128 * 32, dec_rb1 + 128, BUFD, BUFB);
  // convT1: 128->64, 4x4 s2 p1, 32x32 -> 64x64, ReLU(in)
  convt_wmma<128, 64, 32, 32, 64, 64>
      <<<dim3(64 * 64 * 4), dim3(128), 0, stream>>>(BUFB, WPT1, dec_tb1, BUFA);
  dec_convt2<<<dim3(12288), dim3(256), 0, stream>>>(BUFA, dec_tw2, dec_tb2, out);

  // ---- losses (out_dec is 64*3*128*128 = 3145728 floats) ----
  finalize_losses<<<dim3(1), dim3(1), 0, stream>>>(SUM, out + 3145728);
}
